// MambaBlock_62277025792204
// MI455X (gfx1250) — matrix operand order repaired and checked by reading, not compiled, not run on hardware
//
#include <hip/hip_runtime.h>
#include <math.h>

// ---------------------------------------------------------------------------
// Problem constants (match reference)
// ---------------------------------------------------------------------------
#define BB    4
#define LL    2048
#define LC    2051          // L + d_conv - 1
#define DM    1024          // d_model
#define DIMC  2048          // expand * d_model == d_state
#define NX    6144          // 3 * d_state
#define DCONV 4

#define M1 ((size_t)BB * LL)   // 8192 rows into in_proj
#define M2 ((size_t)BB * LC)   // 8204 rows after conv

// ---------------------------------------------------------------------------
// Vector types for WMMA
// ---------------------------------------------------------------------------
typedef __attribute__((ext_vector_type(8)))  __bf16 v8bf;
typedef __attribute__((ext_vector_type(16))) __bf16 v16bf;
typedef __attribute__((ext_vector_type(8)))  float  v8f;

// ---------------------------------------------------------------------------
// CDNA5 async global->LDS copy (ASYNCcnt-tracked; ISA 15.18.3 op 98).
// lds_off: byte offset within the wave's LDS allocation (generic addr [31:0]).
// ---------------------------------------------------------------------------
__device__ __forceinline__ void async_ld_b128(unsigned lds_off, const void* gaddr) {
    asm volatile("global_load_async_to_lds_b128 %0, %1, off"
                 :: "v"(lds_off), "v"(gaddr)
                 : "memory");
}

// ---------------------------------------------------------------------------
// fp32 -> bf16 conversion kernels
// ---------------------------------------------------------------------------
__global__ void cvt_f32_bf16_kernel(const float* __restrict__ src,
                                    __bf16* __restrict__ dst, long n) {
    long i = (long)blockIdx.x * blockDim.x + threadIdx.x;
    if (i < n) dst[i] = (__bf16)src[i];
}

// strided slice (delta_raw = x_dbl[:, 0:DIMC]) -> bf16
__global__ void cvt_slice_bf16_kernel(const float* __restrict__ src,
                                      __bf16* __restrict__ dst,
                                      long rows, int cols, int src_stride) {
    long i = (long)blockIdx.x * blockDim.x + threadIdx.x;
    long n = rows * (long)cols;
    if (i < n) {
        long r = i / cols;
        int  c = (int)(i - r * cols);
        dst[i] = (__bf16)src[r * (size_t)src_stride + c];
    }
}

// ---------------------------------------------------------------------------
// bf16 WMMA GEMM:  C[M,N] (f32, row-major) = A[M,K] @ W[N,K]^T + bias, opt softplus
// 256 threads = 8 waves; block tile 128x128, K tile 32; wave tile 32x64.
// Double-buffered LDS fed by global_load_async_to_lds_b128 (ASYNCcnt),
// one-tile-deep software pipeline: prefetch k+1 while WMMAs run on k.
// ---------------------------------------------------------------------------
#define TM   128
#define TN   128
#define TK   32
#define LDSW 40                    // bf16 elems per LDS row (32 + 8 pad), 80B row
#define ABUF_BYTES (TM * LDSW * 2) // 10240 bytes per A buffer (B identical)

__global__ __launch_bounds__(256)
void gemm_bf16_wmma_kernel(const __bf16* __restrict__ A,
                           const __bf16* __restrict__ W,
                           const float*  __restrict__ bias,
                           float* __restrict__ C,
                           int M, int N, int K, int softplus_act)
{
    __shared__ __bf16 sA[2][TM * LDSW];
    __shared__ __bf16 sB[2][TN * LDSW];

    const int tid  = threadIdx.x;
    const int lane = tid & 31;
    const int wv   = tid >> 5;     // 0..7
    const int wm   = wv & 3;       // 4 waves along M (32 rows each)
    const int wn   = wv >> 2;      // 2 waves along N (64 cols each)
    const int hlf  = lane >> 4;    // 0/1
    const int l16  = lane & 15;

    const int bm0 = blockIdx.x * TM;
    const int bn0 = blockIdx.y * TN;

    // Staging descriptors: 2 x 16B chunks per matrix per thread.
    // Row indices are CLAMPED (never OOB): a duplicated A row only affects
    // that same output row, which is masked at the epilogue store.
    const __bf16* ga[2];
    const __bf16* gb[2];
    unsigned la0[2], lb0[2];       // LDS byte offsets into buffer 0
#pragma unroll
    for (int jj = 0; jj < 2; ++jj) {
        int chunk = tid + jj * 256;
        int row   = chunk >> 2;
        int c8    = (chunk & 3) << 3;
        int gm = bm0 + row; if (gm > M - 1) gm = M - 1;
        int gn = bn0 + row; if (gn > N - 1) gn = N - 1;
        ga[jj] = A + (size_t)gm * K + c8;
        gb[jj] = W + (size_t)gn * K + c8;
        la0[jj] = (unsigned)(size_t)&sA[0][row * LDSW + c8];
        lb0[jj] = (unsigned)(size_t)&sB[0][row * LDSW + c8];
    }

    v8f acc[2][4];
#pragma unroll
    for (int i = 0; i < 2; ++i)
#pragma unroll
        for (int j = 0; j < 4; ++j)
#pragma unroll
            for (int r = 0; r < 8; ++r) acc[i][j][r] = 0.0f;

    const int nk = K / TK;         // K is always a multiple of 32 here

    // Prologue: stage tile 0 into buffer 0 (4 async b128 per wave).
#pragma unroll
    for (int jj = 0; jj < 2; ++jj) {
        async_ld_b128(la0[jj], ga[jj]);
        async_ld_b128(lb0[jj], gb[jj]);
        ga[jj] += TK;
        gb[jj] += TK;
    }

    int buf = 0;
    for (int kt = 0; kt < nk; ++kt) {
        __syncthreads();                       // all readers of buf^1 are done
        if (kt + 1 < nk) {
            unsigned doff = (unsigned)(buf ^ 1) * ABUF_BYTES;
#pragma unroll
            for (int jj = 0; jj < 2; ++jj) {   // prefetch tile kt+1
                async_ld_b128(la0[jj] + doff, ga[jj]);
                async_ld_b128(lb0[jj] + doff, gb[jj]);
                ga[jj] += TK;
                gb[jj] += TK;
            }
            // 8 outstanding; wait until <=4 -> tile kt has landed (in-order)
            asm volatile("s_wait_asynccnt 0x4" ::: "memory");
        } else {
            asm volatile("s_wait_asynccnt 0x0" ::: "memory");
        }
        __syncthreads();                       // tile kt visible to all waves

        const __bf16* bufA = &sA[0][0] + (size_t)buf * (TM * LDSW);
        const __bf16* bufB = &sB[0][0] + (size_t)buf * (TN * LDSW);

        // A fragment (16x32): lane holds K = {hlf*8..+7} U {16+hlf*8..+7}
        v16bf af[2];
#pragma unroll
        for (int i = 0; i < 2; ++i) {
            const __bf16* p = bufA + (wm * 32 + i * 16 + l16) * LDSW + hlf * 8;
            v8bf lo = *(const v8bf*)p;
            v8bf hi = *(const v8bf*)(p + 16);
            af[i] = __builtin_shufflevector(lo, hi, 0,1,2,3,4,5,6,7,
                                                    8,9,10,11,12,13,14,15);
        }
        // B fragment (32x16): lane (col = l16) holds K = {hlf*8..+7} U {16+hlf*8..+7}
        v16bf bf_[4];
#pragma unroll
        for (int j = 0; j < 4; ++j) {
            const __bf16* p = bufB + (wn * 64 + j * 16 + l16) * LDSW + hlf * 8;
            v8bf lo = *(const v8bf*)p;
            v8bf hi = *(const v8bf*)(p + 16);
            bf_[j] = __builtin_shufflevector(lo, hi, 0,1,2,3,4,5,6,7,
                                                     8,9,10,11,12,13,14,15);
        }
#pragma unroll
        for (int i = 0; i < 2; ++i)
#pragma unroll
            for (int j = 0; j < 4; ++j)
                acc[i][j] = __builtin_amdgcn_wmma_f32_16x16x32_bf16(
                    false, af[i], false, bf_[j], (short)0, acc[i][j],
                    false, false);
        buf ^= 1;
    }

    // Epilogue: bias (+ optional softplus), bounds-checked rows
#pragma unroll
    for (int i = 0; i < 2; ++i)
#pragma unroll
        for (int j = 0; j < 4; ++j) {
            int   col = bn0 + wn * 64 + j * 16 + l16;
            float bv  = bias ? bias[col] : 0.0f;
#pragma unroll
            for (int r = 0; r < 8; ++r) {
                int rowg = bm0 + wm * 32 + i * 16 + hlf * 8 + r;
                if (rowg < M) {
                    float v = acc[i][j][r] + bv;
                    if (softplus_act) v = (v > 20.0f) ? v : log1pf(__expf(v));
                    C[(size_t)rowg * N + col] = v;
                }
            }
        }
}

// ---------------------------------------------------------------------------
// Depthwise conv1d, torch padding = d_conv-1 both sides -> length LC = L+3
// Writes fp32 (for scan) and bf16 (for x_proj GEMM).
// ---------------------------------------------------------------------------
__global__ void dwconv_kernel(const float* __restrict__ xp,
                              const float* __restrict__ cw,
                              const float* __restrict__ cb,
                              float* __restrict__ xc,
                              __bf16* __restrict__ xcbf)
{
    long idx = (long)blockIdx.x * blockDim.x + threadIdx.x;
    if (idx >= (long)M2 * DIMC) return;
    int  c   = (int)(idx % DIMC);
    long row = idx / DIMC;            // b*LC + t
    int  t   = (int)(row % LC);
    int  b   = (int)(row / LC);
    float a = cb[c];
#pragma unroll
    for (int k = 0; k < DCONV; ++k) {
        int ti = t - (DCONV - 1) + k;
        if (ti >= 0 && ti < LL)
            a += xp[((size_t)b * LL + ti) * DIMC + c] * cw[c * DCONV + k];
    }
    xc[idx]   = a;
    xcbf[idx] = (__bf16)a;
}

// ---------------------------------------------------------------------------
// Channelwise selective scan: one thread per (batch, channel); fully
// coalesced strided loads across the wave; 8192 independent recurrences.
// B/C read as strided slices of x_dbl (cols [DIMC,2*DIMC) and [2*DIMC,3*DIMC)).
// ---------------------------------------------------------------------------
__global__ void scan_kernel(const float* __restrict__ delta,
                            const float* __restrict__ xc,
                            const float* __restrict__ xdbl,
                            float* __restrict__ ys,
                            __bf16* __restrict__ ysbf)
{
    int idx = blockIdx.x * blockDim.x + threadIdx.x;
    if (idx >= BB * DIMC) return;
    int c = idx & (DIMC - 1);
    int b = idx / DIMC;
    const float Ac = -__expf(-5.0f * (float)c / (float)(DIMC - 1));
    float h = 0.0f;
    for (int t = 0; t < LC; ++t) {
        size_t r  = (size_t)b * LC + t;
        float  d  = delta[r * DIMC + c];
        float  xv = xc[r * DIMC + c];
        float  bm = xdbl[r * NX + DIMC + c];
        float  cm = xdbl[r * NX + 2 * DIMC + c];
        h = h * __expf(d * Ac) + xv * bm;
        float y = h * cm;
        ys[r * DIMC + c]   = y;
        ysbf[r * DIMC + c] = (__bf16)y;
    }
}

// ---------------------------------------------------------------------------
// Host launcher
// ---------------------------------------------------------------------------
extern "C" void kernel_launch(void* const* d_in, const int* in_sizes, int n_in,
                              void* d_out, int out_size, void* d_ws, size_t ws_size,
                              hipStream_t stream)
{
    const float* x    = (const float*)d_in[0];
    const float* inw  = (const float*)d_in[1];
    const float* inb  = (const float*)d_in[2];
    const float* cw   = (const float*)d_in[3];
    const float* cb   = (const float*)d_in[4];
    const float* xpw  = (const float*)d_in[5];
    const float* dtw  = (const float*)d_in[6];
    const float* dtb  = (const float*)d_in[7];
    const float* ow   = (const float*)d_in[8];
    const float* ob   = (const float*)d_in[9];

    char*  ws  = (char*)d_ws;
    size_t off = 0;
    auto alloc = [&](size_t bytes) -> char* {
        char* p = ws + off;
        off = (off + bytes + 255) & ~(size_t)255;
        return p;
    };

    __bf16* w_in  = (__bf16*)alloc((size_t)DIMC * DM   * 2);
    __bf16* w_xp  = (__bf16*)alloc((size_t)NX   * DIMC * 2);
    __bf16* w_dt  = (__bf16*)alloc((size_t)DIMC * DIMC * 2);
    __bf16* w_out = (__bf16*)alloc((size_t)DM   * DIMC * 2);
    __bf16* x_bf  = (__bf16*)alloc(M1 * DM   * 2);
    // Region reused: first as xp (in_proj out, M1 rows), later as delta (M2 rows)
    float*  xp_delta = (float*)alloc(M2 * DIMC * 4);
    float*  xc    = (float*) alloc(M2 * DIMC * 4);
    __bf16* xc_bf = (__bf16*)alloc(M2 * DIMC * 2);
    float*  xdbl  = (float*) alloc(M2 * (size_t)NX * 4);
    __bf16* dr_bf = (__bf16*)alloc(M2 * DIMC * 2);
    float*  ysf   = (float*) alloc(M2 * DIMC * 4);
    __bf16* ys_bf = (__bf16*)alloc(M2 * DIMC * 2);

    const int T = 256;
    auto blks = [](long n) { return (unsigned)((n + 255) / 256); };

    // Cast activations + weights to bf16
    cvt_f32_bf16_kernel<<<blks((long)M1 * DM),       T, 0, stream>>>(x,   x_bf,  (long)M1 * DM);
    cvt_f32_bf16_kernel<<<blks((long)DIMC * DM),     T, 0, stream>>>(inw, w_in,  (long)DIMC * DM);
    cvt_f32_bf16_kernel<<<blks((long)NX * DIMC),     T, 0, stream>>>(xpw, w_xp,  (long)NX * DIMC);
    cvt_f32_bf16_kernel<<<blks((long)DIMC * DIMC),   T, 0, stream>>>(dtw, w_dt,  (long)DIMC * DIMC);
    cvt_f32_bf16_kernel<<<blks((long)DM * DIMC),     T, 0, stream>>>(ow,  w_out, (long)DM * DIMC);

    // 1) in_proj: xp = x @ in_w^T + in_b           (8192 x 1024 -> 2048)
    {
        dim3 g((unsigned)((M1 + TM - 1) / TM), (unsigned)(DIMC / TN));
        gemm_bf16_wmma_kernel<<<g, 256, 0, stream>>>(x_bf, w_in, inb, xp_delta,
                                                     (int)M1, DIMC, DM, 0);
    }
    // 2) depthwise conv -> xc (fp32 + bf16)
    dwconv_kernel<<<blks((long)M2 * DIMC), T, 0, stream>>>(xp_delta, cw, cb, xc, xc_bf);

    // 3) x_proj: x_dbl = xc @ xproj_w^T            (8204 x 2048 -> 6144)
    {
        dim3 g((unsigned)((M2 + TM - 1) / TM), (unsigned)(NX / TN));
        gemm_bf16_wmma_kernel<<<g, 256, 0, stream>>>(xc_bf, w_xp, nullptr, xdbl,
                                                     (int)M2, NX, DIMC, 0);
    }
    // 4) delta_raw slice -> bf16
    cvt_slice_bf16_kernel<<<blks((long)M2 * DIMC), T, 0, stream>>>(xdbl, dr_bf,
                                                                   (long)M2, DIMC, NX);
    // 5) dt_proj + softplus: delta (reuses xp buffer) (8204 x 2048 -> 2048)
    {
        dim3 g((unsigned)((M2 + TM - 1) / TM), (unsigned)(DIMC / TN));
        gemm_bf16_wmma_kernel<<<g, 256, 0, stream>>>(dr_bf, w_dt, dtb, xp_delta,
                                                     (int)M2, DIMC, DIMC, 1);
    }
    // 6) selective scan -> ys (fp32 + bf16)
    scan_kernel<<<(BB * DIMC) / T, T, 0, stream>>>(xp_delta, xc, xdbl, ysf, ys_bf);

    // 7) out_proj: out = ys @ out_w^T + out_b      (8204 x 2048 -> 1024)
    {
        dim3 g((unsigned)((M2 + TM - 1) / TM), (unsigned)(DM / TN));
        gemm_bf16_wmma_kernel<<<g, 256, 0, stream>>>(ys_bf, w_out, ob, (float*)d_out,
                                                     (int)M2, DM, DIMC, 0);
    }
    (void)in_sizes; (void)n_in; (void)out_size; (void)ws_size;
}
